// MPCNN_32444182954387
// MI455X (gfx1250) — matrix operand order repaired and checked by reading, not compiled, not run on hardware
//
#include <hip/hip_runtime.h>
#include <hip/hip_bf16.h>
#include <math.h>

// ---------------- problem constants ----------------
#define BB     128
#define LL     64
#define DD     300
#define NHH    300
#define NPDD   20
#define HIDD   150
#define NFEAT  44427
#define NFEATP 44448      // padded to multiple of 32 for bf16 WMMA K-loop
#define NHP    304        // NH padded to 19*16
#define LP     64
#define HIDP   160        // HID padded to 10*16
#define TP     80         // padded time axis for bf16 X (covers t+kw <= 65)
#define DP     320        // padded D axis (10 K-blocks of 32)

typedef unsigned short ushort_t;

// ---------------- workspace layout (floats) ----------------
#define OFF_SENT  0ull
#define SZ_SENT   (2ull*BB*DD*LP)
#define OFF_HBUF  (OFF_SENT + SZ_SENT)
#define SZ_HBUF   ((size_t)BB*NHP*LP)
#define OFF_AMAX  (OFF_HBUF + SZ_HBUF)
#define SZ_A      (2ull*3*BB*NHH)
#define OFF_ANEG  (OFF_AMAX + SZ_A)
#define OFF_AMEAN (OFF_ANEG + SZ_A)
#define OFF_INF   (OFF_AMEAN + SZ_A)
#define SZ_INF    (2ull*3*BB)
#define OFF_PMAX  (OFF_INF + SZ_INF)
#define SZ_P      (2ull*3*BB*DD*NPDD)
#define OFF_PNEG  (OFF_PMAX + SZ_P)
#define OFF_FEAT  (OFF_PNEG + SZ_P)
#define SZ_FEAT   ((size_t)BB*NFEATP)
#define OFF_HID   (OFF_FEAT + SZ_FEAT)
#define SZ_HID    ((size_t)BB*HIDP)
// bf16 regions (sizes given in floats; 2 ushorts per float; all 16B-multiples)
#define OFF_SBF   (OFF_HID + SZ_HID)            // X bf16, transposed: [s][b][TP][DP]
#define SZ_SBF    ((size_t)BB*TP*DP)            // 2*BB*TP*DP ushorts
#define OFF_WPK   (OFF_SBF + SZ_SBF)            // W bf16: 6 slabs [kw][NHP][DP]
#define SZ_WPK    (6ull*NHP*DP/2)
#define OFF_LW1T  (OFF_WPK + SZ_WPK)            // lw1^T bf16: [HIDP][NFEATP]
#define SZ_LW1T   ((size_t)HIDP*NFEATP/2)
#define OFF_FBF   (OFF_LW1T + SZ_LW1T)          // feat bf16: [BB][NFEATP]
#define SZ_FBF    ((size_t)BB*NFEATP/2)

typedef __attribute__((ext_vector_type(16))) __bf16    v16bf;
typedef __attribute__((ext_vector_type(8)))  float     v8f;
typedef __attribute__((ext_vector_type(4)))  unsigned  v4u;
typedef __attribute__((ext_vector_type(8)))  unsigned  v8u;

__device__ __forceinline__ ushort_t f2bfu(float f) {
  unsigned u = __builtin_bit_cast(unsigned, f);
  unsigned r = u + 0x7fffu + ((u >> 16) & 1u);      // round-to-nearest-even
  return (ushort_t)(r >> 16);
}

// Build a WMMA bf16 fragment from two contiguous 16B runs (no repacking VALU).
__device__ __forceinline__ v16bf frag2(const ushort_t* p0, const ushort_t* p1) {
  v4u a = *(const v4u*)p0;
  v4u b = *(const v4u*)p1;
  v8u w = __builtin_shufflevector(a, b, 0, 1, 2, 3, 4, 5, 6, 7);
  return __builtin_bit_cast(v16bf, w);
}

// ---------------- 0) zero the padded bf16 X region ----------------
__global__ void mp_zerobf(float* __restrict__ ws) {
  unsigned* p = (unsigned*)(ws + OFF_SBF);        // SZ_SBF dwords
  size_t n = SZ_SBF;
  for (size_t i = blockIdx.x * (size_t)blockDim.x + threadIdx.x; i < n;
       i += (size_t)gridDim.x * blockDim.x)
    p[i] = 0u;
}

// ---------------- 1) embedding gather -> sent fp32 + transposed bf16 ----------------
__global__ void mp_gather(const int* __restrict__ idx1, const int* __restrict__ idx2,
                          const float* __restrict__ emb, float* __restrict__ ws) {
  ushort_t* xbf = (ushort_t*)(ws + OFF_SBF);
  size_t n = 2ull * BB * LL * DD;
  for (size_t i = blockIdx.x * (size_t)blockDim.x + threadIdx.x; i < n;
       i += (size_t)gridDim.x * blockDim.x) {
    int d = (int)(i % DD);
    size_t r = i / DD;
    int t = (int)(r % LL);
    size_t r2 = r / LL;
    int b = (int)(r2 % BB);
    int s = (int)(r2 / BB);
    const int* idx = s ? idx2 : idx1;
    int v = idx[b * LL + t];
    float e = emb[(size_t)v * DD + d];
    ws[OFF_SENT + ((((size_t)s * BB + b) * DD + d) * LP + t)] = e;
    xbf[(((size_t)(s * BB + b)) * TP + t) * DP + d] = f2bfu(e);   // transposed [t][d]
  }
}

// ---------------- 2) pack conv weights -> bf16 slabs [kw][NHP][DP] ----------------
__global__ void mp_packw(const float* __restrict__ w, float* __restrict__ ws,
                         int slab0, int wsz) {
  ushort_t* wpk = (ushort_t*)(ws + OFF_WPK);
  int total = wsz * NHP * DP;
  for (int i = blockIdx.x * blockDim.x + threadIdx.x; i < total;
       i += gridDim.x * blockDim.x) {
    int kw = i / (NHP * DP);
    int rem = i % (NHP * DP);
    int m = rem / DP, dk = rem % DP;
    float v = (m < NHH && dk < DD) ? w[((size_t)m * DD + dk) * wsz + kw] : 0.f;
    wpk[(size_t)(slab0 + kw) * NHP * DP + (size_t)m * DP + dk] = f2bfu(v);
  }
}

// ---------------- 3) pack lw1 -> transposed bf16 [HIDP][NFEATP] ----------------
__global__ void mp_packlw1(const float* __restrict__ lw1, float* __restrict__ ws) {
  ushort_t* t = (ushort_t*)(ws + OFF_LW1T);
  size_t total = (size_t)HIDP * NFEATP;
  for (size_t i = blockIdx.x * (size_t)blockDim.x + threadIdx.x; i < total;
       i += (size_t)gridDim.x * blockDim.x) {
    int n = (int)(i / NFEATP), k = (int)(i % NFEATP);
    float v = (n < HIDD && k < NFEAT) ? lw1[(size_t)k * HIDD + n] : 0.f;
    t[i] = f2bfu(v);
  }
}

// ---------------- 4) 'inf' block: flat max / (-x).max / mean ----------------
__global__ void mp_infblock(float* __restrict__ ws) {
  int i = blockIdx.x * blockDim.x + threadIdx.x;
  if (i >= 2 * BB) return;
  int b = i % BB, s = i / BB;
  const float* x = ws + OFF_SENT + ((size_t)s * BB + b) * DD * LP;
  float mx = -1e30f, mn = -1e30f, sm = 0.f;
  for (int d = 0; d < DD; ++d)
    for (int t = 0; t < LL; ++t) {
      float v = x[(size_t)d * LP + t];
      mx = fmaxf(mx, v); mn = fmaxf(mn, -v); sm += v;
    }
  ws[OFF_INF + (size_t)(s * 3 + 0) * BB + b] = mx;
  ws[OFF_INF + (size_t)(s * 3 + 1) * BB + b] = mn;
  ws[OFF_INF + (size_t)(s * 3 + 2) * BB + b] = sm / (float)(DD * LL);
}

// ---------------- 5) holistic conv: async-LDS staged bf16 WMMA ----------------
// Per block: stage this batch-row's X slab (TP x DP bf16 = 50KB) into LDS with
// global_load_async_to_lds_b128, then 8 waves each compute one 16x16 tile of
// C[304 x 64] = sum_kw W_kw @ X[:, kw:kw+T], tanh at store. Inner loop per
// K-step: 2x global_load_b128 (A) + 2x ds_load_b128 (B) + 1x v_wmma.
__global__ void mp_hconv_wmma(const ushort_t* __restrict__ xbf,  // [b][TP][DP] bf16 (this sentence)
                              const ushort_t* __restrict__ wpk,  // [kw][NHP][DP] bf16 (this ws)
                              const float* __restrict__ bias,
                              float* __restrict__ ws, int wsz) {
  __shared__ __align__(16) ushort_t sX[TP * DP];   // 51,200 bytes
  int tid = threadIdx.x;
  int b = blockIdx.y;

  // --- async stage: 3200 x 16B chunks = 12 uniform rounds + half-block tail ---
  {
    unsigned lbase = (unsigned)(size_t)(void*)&sX[0];     // generic->LDS: low 32 bits
    unsigned long long gbase = (unsigned long long)(size_t)(xbf + (size_t)b * TP * DP);
    #pragma unroll 1
    for (int it = 0; it < 12; ++it) {
      unsigned c = (unsigned)(tid + it * 256);
      unsigned loff = lbase + c * 16u;
      unsigned long long ga = gbase + (unsigned long long)c * 16ull;
      asm volatile("global_load_async_to_lds_b128 %0, %1, off"
                   :: "v"(loff), "v"(ga) : "memory");
    }
    if (tid < 128) {
      unsigned c = (unsigned)(3072 + tid);
      unsigned loff = lbase + c * 16u;
      unsigned long long ga = gbase + (unsigned long long)c * 16ull;
      asm volatile("global_load_async_to_lds_b128 %0, %1, off"
                   :: "v"(loff), "v"(ga) : "memory");
    }
    asm volatile("s_wait_asynccnt 0x0" ::: "memory");
  }
  __syncthreads();

  int wave = tid >> 5, lane = tid & 31;
  int tile = blockIdx.x * 8 + wave;
  if (tile < 19 * 4) {                             // wave-uniform predicate
    int tm = tile % 19, tn = tile / 19;
    int r = lane & 15, half = lane >> 4;
    float* hbuf = ws + OFF_HBUF;

    v8f acc;
    #pragma unroll
    for (int v = 0; v < 8; ++v) {
      int m = tm * 16 + v + 8 * half;
      acc[v] = (m < NHH) ? bias[m] : 0.f;          // conv bias folded into accumulator
    }

    const ushort_t* arow = wpk + (size_t)(tm * 16 + r) * DP + half * 8;  // A row base
    for (int kw = 0; kw < wsz; ++kw) {
      int t_in = tn * 16 + r + kw;                 // B column (shifted), always < TP
      const ushort_t* aw = arow + (size_t)kw * NHP * DP;
      const ushort_t* bx = &sX[(size_t)t_in * DP + half * 8];
      for (int kb = 0; kb < 10; ++kb) {
        // A 16x32 bf16 fragment: two contiguous 16B runs (k0 = half*8 + 2v, +16)
        v16bf A = frag2(aw + kb * 32, aw + kb * 32 + 16);
        // B 32x16 bf16 fragment: two contiguous 16B runs (k0 = half*8 + 2v, +16)
        v16bf Bm = frag2(bx + kb * 32, bx + kb * 32 + 16);
        acc = __builtin_amdgcn_wmma_f32_16x16x32_bf16(false, A, false, Bm,
                                                      (short)0, acc, false, false);
      }
    }
    #pragma unroll
    for (int v = 0; v < 8; ++v) {
      int m = tm * 16 + v + 8 * half;              // C/D layout: M = v + 8*(lane/16)
      int t = tn * 16 + r;                         // N = lane%16
      hbuf[((size_t)b * NHP + m) * LP + t] = tanhf(acc[v]);
    }
  }
}

// ---------------- 6) holistic pooling over valid T ----------------
__global__ void mp_hpool(float* __restrict__ ws, int s, int wsi, int T) {
  int i = blockIdx.x * blockDim.x + threadIdx.x;
  if (i >= BB * NHH) return;
  int b = i / NHH, nh = i % NHH;
  const float* h = ws + OFF_HBUF + ((size_t)b * NHP + nh) * LP;
  float mx = -1e30f, mn = -1e30f, sm = 0.f;
  for (int t = 0; t < T; ++t) {
    float v = h[t];
    mx = fmaxf(mx, v); mn = fmaxf(mn, -v); sm += v;
  }
  size_t o = ((size_t)(s * 3 + wsi) * BB + b) * NHH + nh;
  ws[OFF_AMAX + o] = mx; ws[OFF_ANEG + o] = mn; ws[OFF_AMEAN + o] = sm / (float)T;
}

// ---------------- 7) depthwise (per-dim) conv + pooling, fused ----------------
__global__ void mp_pconv(const float* __restrict__ sent, const float* __restrict__ pw,
                         const float* __restrict__ pb, float* __restrict__ ws,
                         int s, int wsi, int wsz, int T) {
  int i = blockIdx.x * blockDim.x + threadIdx.x;
  if (i >= BB * DD * NPDD) return;
  int c = i % (DD * NPDD), b = i / (DD * NPDD), d = c / NPDD;
  const float* x = sent + ((size_t)b * DD + d) * LP;
  float mx = -1e30f, mn = -1e30f;
  for (int t = 0; t < T; ++t) {
    float acc = pb[c];
    for (int k = 0; k < wsz; ++k) acc += pw[c * wsz + k] * x[t + k];
    float v = tanhf(acc);
    mx = fmaxf(mx, v); mn = fmaxf(mn, -v);
  }
  size_t o = ((size_t)(s * 3 + wsi) * BB + b) * DD * NPDD + c;
  ws[OFF_PMAX + o] = mx; ws[OFF_PNEG + o] = mn;
}

// ---------------- 8) feature builder: one wave per batch row ----------------
__device__ __forceinline__ float wsum32(float v) {
  #pragma unroll
  for (int o = 16; o > 0; o >>= 1) v += __shfl_xor(v, o, 32);
  return v;
}

__device__ void mp_cosdist(const float* x, int sx, const float* y, int sy, int n,
                           int lane, float* pc, float* pd) {
  float dot = 0.f, nx = 0.f, ny = 0.f, dd = 0.f;
  for (int i = lane; i < n; i += 32) {
    float a = x[(size_t)i * sx], b = y[(size_t)i * sy];
    dot += a * b; nx += a * a; ny += b * b;
    float df = a - b + 1e-6f; dd += df * df;
  }
  dot = wsum32(dot); nx = wsum32(nx); ny = wsum32(ny); dd = wsum32(dd);
  *pc = dot / (fmaxf(sqrtf(nx), 1e-8f) * fmaxf(sqrtf(ny), 1e-8f));
  *pd = sqrtf(dd);
}

__device__ const float* mp_avec(float* ws, int s, int pool, int w, int b, int* n) {
  if (w < 3) {
    *n = NHH;
    size_t base = (pool == 0) ? OFF_AMAX : (pool == 1) ? OFF_ANEG : OFF_AMEAN;
    return ws + base + ((size_t)(s * 3 + w) * BB + b) * NHH;
  }
  *n = 1;
  return ws + OFF_INF + (size_t)(s * 3 + pool) * BB + b;
}

__global__ void mp_features(float* __restrict__ ws) {
  int b = blockIdx.x, lane = threadIdx.x;
  float* feat = ws + OFF_FEAT + (size_t)b * NFEATP;
  int idx = 0;
  // loop 1: same (pool, ws) both sides -> cos, dist
  for (int pool = 0; pool < 3; ++pool)
    for (int w = 0; w < 4; ++w) {
      int n; const float* x1 = mp_avec(ws, 0, pool, w, b, &n);
      const float* x2 = mp_avec(ws, 1, pool, w, b, &n);
      float c, d; mp_cosdist(x1, 1, x2, 1, n, lane, &c, &d);
      if (lane == 0) { feat[idx] = c; feat[idx + 1] = d; }
      idx += 2;
    }
  // loop 2: cross-(ws1, ws2), both-finite or both-inf -> cos, dist, |x1-x2|
  for (int pool = 0; pool < 3; ++pool)
    for (int w1 = 0; w1 < 4; ++w1)
      for (int w2 = 0; w2 < 4; ++w2) {
        bool f1 = w1 < 3, f2 = w2 < 3;
        if ((f1 && f2) || (!f1 && !f2)) {
          int n; const float* x1 = mp_avec(ws, 0, pool, w1, b, &n);
          const float* x2 = mp_avec(ws, 1, pool, w2, b, &n);
          float c, d; mp_cosdist(x1, 1, x2, 1, n, lane, &c, &d);
          if (lane == 0) { feat[idx] = c; feat[idx + 1] = d; }
          idx += 2;
          for (int i = lane; i < n; i += 32) feat[idx + i] = fabsf(x1[i] - x2[i]);
          idx += n;
        }
      }
  // loop 3: per-dim blocks, pools {max, min}, vectors over D with stride NPD
  for (int pool = 0; pool < 2; ++pool)
    for (int wsi = 0; wsi < 3; ++wsi)
      for (int i = 0; i < NPDD; ++i) {
        size_t base = (pool == 0) ? OFF_PMAX : OFF_PNEG;
        const float* x1 = ws + base + ((size_t)(0 * 3 + wsi) * BB + b) * DD * NPDD + i;
        const float* x2 = ws + base + ((size_t)(1 * 3 + wsi) * BB + b) * DD * NPDD + i;
        float c, d; mp_cosdist(x1, NPDD, x2, NPDD, DD, lane, &c, &d);
        if (lane == 0) { feat[idx] = c; feat[idx + 1] = d; }
        idx += 2;
        for (int j = lane; j < DD; j += 32)
          feat[idx + j] = fabsf(x1[(size_t)j * NPDD] - x2[(size_t)j * NPDD]);
        idx += DD;
      }
  // zero-pad tail so the bf16 WMMA K-loop sees zeros
  for (int i = idx + lane; i < NFEATP; i += 32) feat[i] = 0.f;
}

// ---------------- 9) pack features to bf16 ----------------
__global__ void mp_packfeat(float* __restrict__ ws) {
  ushort_t* fb = (ushort_t*)(ws + OFF_FBF);
  const float* f = ws + OFF_FEAT;
  size_t total = (size_t)BB * NFEATP;
  for (size_t i = blockIdx.x * (size_t)blockDim.x + threadIdx.x; i < total;
       i += (size_t)gridDim.x * blockDim.x)
    fb[i] = f2bfu(f[i]);
}

// ---------------- 10) MLP layer 1 via bf16 WMMA (pre-packed, branch-free) ----------------
__global__ void mp_mlp_wmma(const float* __restrict__ lb1, float* __restrict__ ws) {
  const ushort_t* fb   = (const ushort_t*)(ws + OFF_FBF);    // [BB][NFEATP]
  const ushort_t* lw1t = (const ushort_t*)(ws + OFF_LW1T);   // [HIDP][NFEATP]
  float* hid = ws + OFF_HID;

  int wave = threadIdx.x >> 5, lane = threadIdx.x & 31;
  int tile = blockIdx.x * 8 + wave;            // 80 tiles = 10 blocks * 8 waves
  int tm = tile & 7, tn = tile >> 3;
  int r = lane & 15, half = lane >> 4;

  v8f acc;
  #pragma unroll
  for (int v = 0; v < 8; ++v) acc[v] = 0.f;

  int m = tm * 16 + r;                          // A row (< 128)
  int n = tn * 16 + r;                          // B column (< 160, padded rows are zero)
  const ushort_t* arow = fb   + (size_t)m * NFEATP + half * 8;
  const ushort_t* brow = lw1t + (size_t)n * NFEATP + half * 8;

  for (int kb = 0; kb < NFEATP / 32; ++kb) {
    v16bf A  = frag2(arow + kb * 32, arow + kb * 32 + 16);  // two 16B runs
    v16bf Bm = frag2(brow + kb * 32, brow + kb * 32 + 16);  // two 16B runs
    if (kb + 1 < NFEATP / 32)                   // prefetch next K-slab of streaming operand
      __builtin_prefetch(&brow[(kb + 1) * 32], 0, 1);
    acc = __builtin_amdgcn_wmma_f32_16x16x32_bf16(false, A, false, Bm,
                                                  (short)0, acc, false, false);
  }

  #pragma unroll
  for (int v = 0; v < 8; ++v) {
    int mr = tm * 16 + v + 8 * half;
    int nc = tn * 16 + r;
    if (nc < HIDD) hid[(size_t)mr * HIDP + nc] = tanhf(acc[v] + lb1[nc]);
  }
}

// ---------------- 11) logits + log_softmax ----------------
__global__ void mp_head(const float* __restrict__ lw2, const float* __restrict__ lb2,
                        float* __restrict__ out, const float* __restrict__ wsro) {
  int b = blockIdx.x * blockDim.x + threadIdx.x;
  if (b >= BB) return;
  const float* h = wsro + OFF_HID + (size_t)b * HIDP;
  float l0 = lb2[0], l1 = lb2[1];
  for (int j = 0; j < HIDD; ++j) {
    float v = h[j];
    l0 += v * lw2[j * 2 + 0];
    l1 += v * lw2[j * 2 + 1];
  }
  float m = fmaxf(l0, l1);
  float lse = m + logf(expf(l0 - m) + expf(l1 - m));
  out[b * 2 + 0] = l0 - lse;
  out[b * 2 + 1] = l1 - lse;
}

// ---------------- host-side launch ----------------
extern "C" void kernel_launch(void* const* d_in, const int* in_sizes, int n_in,
                              void* d_out, int out_size, void* d_ws, size_t ws_size,
                              hipStream_t stream) {
  const int*   idx1 = (const int*)d_in[0];
  const int*   idx2 = (const int*)d_in[1];
  const float* emb  = (const float*)d_in[2];
  const float* lw1  = (const float*)d_in[3];
  const float* lb1  = (const float*)d_in[4];
  const float* lw2  = (const float*)d_in[5];
  const float* lb2  = (const float*)d_in[6];
  const float* hw[3] = {(const float*)d_in[7],  (const float*)d_in[11], (const float*)d_in[15]};
  const float* hb[3] = {(const float*)d_in[8],  (const float*)d_in[12], (const float*)d_in[16]};
  const float* pw[3] = {(const float*)d_in[9],  (const float*)d_in[13], (const float*)d_in[17]};
  const float* pb[3] = {(const float*)d_in[10], (const float*)d_in[14], (const float*)d_in[18]};
  float* ws  = (float*)d_ws;
  float* out = (float*)d_out;

  const int slab0[3] = {0, 1, 3};               // W slab prefix per ws

  mp_zerobf<<<dim3(2048), dim3(256), 0, stream>>>(ws);
  mp_gather<<<dim3(2048), dim3(256), 0, stream>>>(idx1, idx2, emb, ws);
  mp_packlw1<<<dim3(4096), dim3(256), 0, stream>>>(lw1, ws);
  for (int wsi = 0; wsi < 3; ++wsi)
    mp_packw<<<dim3(1140), dim3(256), 0, stream>>>(hw[wsi], ws, slab0[wsi], wsi + 1);
  mp_infblock<<<dim3(1), dim3(256), 0, stream>>>(ws);

  const ushort_t* xbf = (const ushort_t*)(ws + OFF_SBF);
  const ushort_t* wpk = (const ushort_t*)(ws + OFF_WPK);
  for (int s = 0; s < 2; ++s) {
    const float* sent = ws + OFF_SENT + (size_t)s * BB * DD * LP;
    const ushort_t* xbf_s = xbf + (size_t)s * BB * TP * DP;
    for (int wsi = 0; wsi < 3; ++wsi) {
      int wsz = wsi + 1, T = LL - wsz + 1;
      const ushort_t* wpk_s = wpk + (size_t)slab0[wsi] * NHP * DP;
      mp_hconv_wmma<<<dim3(10, BB), dim3(256), 0, stream>>>(xbf_s, wpk_s, hb[wsi], ws, wsz);
      mp_hpool<<<dim3((BB * NHH + 255) / 256), dim3(256), 0, stream>>>(ws, s, wsi, T);
      mp_pconv<<<dim3((BB * DD * NPDD + 255) / 256), dim3(256), 0, stream>>>(
          sent, pw[wsi], pb[wsi], ws, s, wsi, wsz, T);
    }
  }

  mp_features<<<dim3(BB), dim3(32), 0, stream>>>(ws);
  mp_packfeat<<<dim3(4096), dim3(256), 0, stream>>>(ws);
  mp_mlp_wmma<<<dim3(10), dim3(256), 0, stream>>>(lb1, ws);
  mp_head<<<dim3(1), dim3(BB), 0, stream>>>(lw2, lb2, out, ws);
}
